// ConditionalPaiNNDecoder_22325240005210
// MI455X (gfx1250) — matrix operand order repaired and checked by reading, not compiled, not run on hardware
//
#include <hip/hip_runtime.h>
#include <hip/hip_bf16.h>
#include <hip/hip_fp16.h>

#define N_NODES 20000
#define N_EDGES 320000
#define HID 128
#define LAYERS 4
#define EDGE_DIM 19

typedef _Float16 h8  __attribute__((ext_vector_type(8)));
typedef _Float16 h16 __attribute__((ext_vector_type(16)));
typedef float    f8  __attribute__((ext_vector_type(8)));
typedef float    f4  __attribute__((ext_vector_type(4)));

__device__ __forceinline__ f8 wmma_f16(h16 a, h16 b, f8 c) {
  return __builtin_amdgcn_wmma_f32_16x16x32_f16(false, a, false, b, (short)0, c, false, false);
}

// A fragment: 16x32 f16, lane = M row (tm), half splits K {0-7,16-23}/{8-15,24-31}
__device__ __forceinline__ h16 frag_a(const _Float16* A, int ld, int row_base, int k0, int lane) {
  int hlf = lane >> 4, tm = lane & 15;
  const _Float16* p = A + (size_t)(row_base + tm) * ld + k0 + hlf * 8;
  h8 lo = *(const h8*)p;
  h8 hi = *(const h8*)(p + 16);
  return __builtin_shufflevector(lo, hi, 0,1,2,3,4,5,6,7,8,9,10,11,12,13,14,15);
}

// B fragment from transposed (NxK) weights: lane = N col (tm), half splits K {0-7,16-23}/{8-15,24-31}
__device__ __forceinline__ h16 frag_b(const _Float16* Bt, int ldk, int col_base, int k0, int lane) {
  int hlf = lane >> 4, tm = lane & 15;
  const _Float16* p = Bt + (size_t)(col_base + tm) * ldk + k0 + hlf * 8;
  h8 lo = *(const h8*)p;
  h8 hi = *(const h8*)(p + 16);
  return __builtin_shufflevector(lo, hi, 0,1,2,3,4,5,6,7,8,9,10,11,12,13,14,15);
}

__device__ __forceinline__ float silu_f(float x) { return x / (1.0f + __expf(-x)); }

// ---------------- init / geometry / weight conversion ----------------

__global__ __launch_bounds__(256) void k_init(const float* __restrict__ atom_latent,
    const float* __restrict__ vecf, const float* __restrict__ embedding,
    const int* __restrict__ z, float* s, _Float16* s_h, float* v, float* v_acc) {
  int id = blockIdx.x * 256 + threadIdx.x;           // over N_NODES*HID
  int n = id >> 7, h = id & 127;
  float sv = embedding[z[n] * HID + h] + atom_latent[id];
  s[id] = sv; s_h[id] = (_Float16)sv;
#pragma unroll
  for (int c = 0; c < 3; ++c) {
    size_t idx = (size_t)n * 384 + c * HID + h;
    v[idx] = vecf[n * 3 + c];
    v_acc[idx] = 0.0f;
  }
}

__global__ __launch_bounds__(256) void k_geom(const float* __restrict__ pos,
    const float* __restrict__ eattr, const int* __restrict__ src, const int* __restrict__ dst,
    float* u, _Float16* ea_h) {
  int e = blockIdx.x * 256 + threadIdx.x;            // over N_EDGES
  int s0 = src[e], d0 = dst[e];
  float rx = pos[d0*3+0] - pos[s0*3+0];
  float ry = pos[d0*3+1] - pos[s0*3+1];
  float rz = pos[d0*3+2] - pos[s0*3+2];
  float dist = sqrtf(rx*rx + ry*ry + rz*rz + 1e-6f);
  u[e*3+0] = rx / dist; u[e*3+1] = ry / dist; u[e*3+2] = rz / dist;
#pragma unroll
  for (int k = 0; k < 32; ++k)
    ea_h[(size_t)e*32 + k] = (k < EDGE_DIM) ? (_Float16)eattr[(size_t)e*EDGE_DIM + k] : (_Float16)0.0f;
}

// transpose-convert f32 (KxN) -> f16 (NxK)
__global__ __launch_bounds__(256) void k_cvt(const float* __restrict__ in, _Float16* out, int K, int N) {
  int id = blockIdx.x * 256 + threadIdx.x;
  if (id >= K * N) return;
  int n = id / K, k = id % K;
  out[id] = (_Float16)in[k * N + n];
}

// filt_W (19x384) -> padded transposed f16 (384x32)
__global__ __launch_bounds__(256) void k_cvt_filt(const float* __restrict__ in, _Float16* out) {
  int id = blockIdx.x * 256 + threadIdx.x;           // over 384*32
  int n = id >> 5, k = id & 31;
  out[id] = (k < EDGE_DIM) ? (_Float16)in[k * 384 + n] : (_Float16)0.0f;
}

// ---------------- node GEMMs ----------------

// Out(f16) = silu(A(20000x128) @ W + b), N=128: wave per 16x16 tile
__global__ __launch_bounds__(256) void k_gemm_silu(const _Float16* __restrict__ Ah,
    const _Float16* __restrict__ Bt, const float* __restrict__ bias, _Float16* Out) {
  int wave = threadIdx.x >> 5, lane = threadIdx.x & 31;
  int hlf = lane >> 4, tm = lane & 15;
  int row = blockIdx.x * 16, col = wave * 16;
  f8 acc = {};
#pragma unroll
  for (int kk = 0; kk < 4; ++kk)
    acc = wmma_f16(frag_a(Ah, HID, row, kk*32, lane), frag_b(Bt, HID, col, kk*32, lane), acc);
  float b = bias[col + tm];
#pragma unroll
  for (int r = 0; r < 8; ++r)
    Out[(size_t)(row + r + 8*hlf) * HID + col + tm] = (_Float16)silu_f(acc[r] + b);
}

// upd1: K=256 split across two f16 sources (s_h, Vn_h), silu epilogue
__global__ __launch_bounds__(256) void k_upd1(const _Float16* __restrict__ Sh,
    const _Float16* __restrict__ Vnh, const _Float16* __restrict__ Bt,
    const float* __restrict__ bias, _Float16* Out) {
  int wave = threadIdx.x >> 5, lane = threadIdx.x & 31;
  int hlf = lane >> 4, tm = lane & 15;
  int row = blockIdx.x * 16, col = wave * 16;
  f8 acc = {};
#pragma unroll
  for (int kk = 0; kk < 8; ++kk) {
    const _Float16* A = (kk < 4) ? Sh : Vnh;
    acc = wmma_f16(frag_a(A, HID, row, (kk & 3) * 32, lane),
                   frag_b(Bt, 256, col, kk * 32, lane), acc);
  }
  float b = bias[col + tm];
#pragma unroll
  for (int r = 0; r < 8; ++r)
    Out[(size_t)(row + r + 8*hlf) * HID + col + tm] = (_Float16)silu_f(acc[r] + b);
}

// phi2: Out(f32, N=384) = A(20000x128) @ W + b ; wave handles 3 col tiles
__global__ __launch_bounds__(256) void k_gemm_n384(const _Float16* __restrict__ Ah,
    const _Float16* __restrict__ Bt, const float* __restrict__ bias, float* Out) {
  int wave = threadIdx.x >> 5, lane = threadIdx.x & 31;
  int hlf = lane >> 4, tm = lane & 15;
  int row = blockIdx.x * 16;
  h16 a[4];
#pragma unroll
  for (int kk = 0; kk < 4; ++kk) a[kk] = frag_a(Ah, HID, row, kk*32, lane);
#pragma unroll
  for (int t = 0; t < 3; ++t) {
    int col = (wave + t * 8) * 16;
    f8 acc = {};
#pragma unroll
    for (int kk = 0; kk < 4; ++kk)
      acc = wmma_f16(a[kk], frag_b(Bt, HID, col, kk*32, lane), acc);
    float b = bias[col + tm];
#pragma unroll
    for (int r = 0; r < 8; ++r)
      Out[(size_t)(row + r + 8*hlf) * 384 + col + tm] = acc[r] + b;
  }
}

// Uv = v @ U_W, Vv = v @ V_W over 60000x128, shared A fragments
__global__ __launch_bounds__(256) void k_uvvv(const _Float16* __restrict__ Vh,
    const _Float16* __restrict__ Ut, const _Float16* __restrict__ Vt,
    float* Uv, float* Vv) {
  int wave = threadIdx.x >> 5, lane = threadIdx.x & 31;
  int hlf = lane >> 4, tm = lane & 15;
  int row = blockIdx.x * 16, col = wave * 16;
  f8 au = {}, av = {};
#pragma unroll
  for (int kk = 0; kk < 4; ++kk) {
    h16 a = frag_a(Vh, HID, row, kk*32, lane);
    au = wmma_f16(a, frag_b(Ut, HID, col, kk*32, lane), au);
    av = wmma_f16(a, frag_b(Vt, HID, col, kk*32, lane), av);
  }
#pragma unroll
  for (int r = 0; r < 8; ++r) {
    size_t idx = (size_t)(row + r + 8*hlf) * HID + col + tm;
    Uv[idx] = au[r]; Vv[idx] = av[r];
  }
}

// ---------------- edge phase: fused filter-GEMM + gather + scatter-add ----------------

__global__ __launch_bounds__(256) void k_edge(const _Float16* __restrict__ EAh,
    const _Float16* __restrict__ FWt, const float* __restrict__ fb,
    const float* __restrict__ phi, const float* __restrict__ v, const float* __restrict__ u,
    const int* __restrict__ src, const int* __restrict__ dst, float* s, float* v_acc) {
  int wave = threadIdx.x >> 5, lane = threadIdx.x & 31;
  int hlf = lane >> 4, tm = lane & 15;
  int ebase = (blockIdx.x * 8 + wave) * 16;
  h16 a = frag_a(EAh, 32, ebase, 0, lane);        // 16 edges x K32 (padded 19)
  int se[8], de[8];
#pragma unroll
  for (int r = 0; r < 8; ++r) {
    int e = ebase + r + 8 * hlf;
    se[r] = src[e]; de[r] = dst[e];
  }
  // scalar messages: cols 0..127 -> atomic add into s[dst]
#pragma unroll
  for (int j = 0; j < 8; ++j) {
    int col = j * 16 + tm;
    f8 z = {};
    f8 d = wmma_f16(a, frag_b(FWt, 32, j * 16, 0, lane), z);
    float b0 = fb[col];
#pragma unroll
    for (int r = 0; r < 8; ++r) {
      float x = phi[(size_t)se[r] * 384 + col] * (d[r] + b0);
      unsafeAtomicAdd(&s[(size_t)de[r] * HID + col], x);
    }
  }
  // vector messages: cols 128..255 (vv) and 256..383 (vd) combined per h
#pragma unroll
  for (int j = 0; j < 8; ++j) {
    int h = j * 16 + tm;
    f8 z1 = {}, z2 = {};
    f8 d1 = wmma_f16(a, frag_b(FWt, 32, 128 + j * 16, 0, lane), z1);
    f8 d2 = wmma_f16(a, frag_b(FWt, 32, 256 + j * 16, 0, lane), z2);
    float b1 = fb[128 + h], b2 = fb[256 + h];
#pragma unroll
    for (int r = 0; r < 8; ++r) {
      int e = ebase + r + 8 * hlf;
      size_t sb = (size_t)se[r] * 384;
      float mvv = phi[sb + 128 + h] * (d1[r] + b1);
      float mvd = phi[sb + 256 + h] * (d2[r] + b2);
#pragma unroll
      for (int c = 0; c < 3; ++c) {
        float contrib = v[sb + c * HID + h] * mvv + u[e * 3 + c] * mvd;
        unsafeAtomicAdd(&v_acc[(size_t)de[r] * 384 + c * HID + h], contrib);
      }
    }
  }
}

// ---------------- per-layer elementwise ----------------

__global__ __launch_bounds__(256) void k_fin(const float* __restrict__ s, _Float16* s_h,
    float* v, float* v_acc, _Float16* v_h) {
  int id = blockIdx.x * 256 + threadIdx.x;           // over N_NODES*HID
  int n = id >> 7, h = id & 127;
  s_h[id] = (_Float16)s[id];
#pragma unroll
  for (int c = 0; c < 3; ++c) {
    size_t idx = (size_t)n * 384 + c * HID + h;
    float nv = v[idx] + v_acc[idx];
    v[idx] = nv; v_h[idx] = (_Float16)nv; v_acc[idx] = 0.0f;
  }
}

__global__ __launch_bounds__(256) void k_vn(const float* __restrict__ Uv,
    const float* __restrict__ Vv, _Float16* Vnh, float* dotv) {
  int id = blockIdx.x * 256 + threadIdx.x;           // over N_NODES*HID
  int n = id >> 7, h = id & 127;
  float ss = 1e-8f, dd = 0.0f;
#pragma unroll
  for (int c = 0; c < 3; ++c) {
    size_t idx = ((size_t)n * 3 + c) * HID + h;
    float uu = Uv[idx], vv = Vv[idx];
    dd += uu * vv; ss += vv * vv;
  }
  Vnh[id] = (_Float16)sqrtf(ss);
  dotv[id] = dd;
}

// upd2 GEMM fused with the s/v update epilogue
__global__ __launch_bounds__(256) void k_upd2(const _Float16* __restrict__ Th,
    const _Float16* __restrict__ Bt, const float* __restrict__ bias,
    const float* __restrict__ Uv, const float* __restrict__ dotv,
    float* v, float* s, _Float16* s_h) {
  int wave = threadIdx.x >> 5, lane = threadIdx.x & 31;
  int hlf = lane >> 4, tm = lane & 15;
  int row = blockIdx.x * 16;
  int hcol = wave * 16 + tm;
  h16 a[4];
#pragma unroll
  for (int kk = 0; kk < 4; ++kk) a[kk] = frag_a(Th, HID, row, kk*32, lane);
  f8 d1 = {}, d2 = {}, d3 = {};
#pragma unroll
  for (int kk = 0; kk < 4; ++kk) d1 = wmma_f16(a[kk], frag_b(Bt, HID, wave*16,        kk*32, lane), d1);
#pragma unroll
  for (int kk = 0; kk < 4; ++kk) d2 = wmma_f16(a[kk], frag_b(Bt, HID, 128 + wave*16,  kk*32, lane), d2);
#pragma unroll
  for (int kk = 0; kk < 4; ++kk) d3 = wmma_f16(a[kk], frag_b(Bt, HID, 256 + wave*16,  kk*32, lane), d3);
  float bvv = bias[hcol], bsv = bias[128 + hcol], bss = bias[256 + hcol];
#pragma unroll
  for (int r = 0; r < 8; ++r) {
    int n = row + r + 8 * hlf;
    float avv = d1[r] + bvv;
#pragma unroll
    for (int c = 0; c < 3; ++c) {
      size_t idx = ((size_t)n * 3 + c) * HID + hcol;
      v[idx] += avv * Uv[idx];
    }
    size_t si = (size_t)n * HID + hcol;
    float ns = s[si] + (d3[r] + bss) + (d2[r] + bsv) * dotv[si];
    s[si] = ns; s_h[si] = (_Float16)ns;
  }
}

__global__ __launch_bounds__(256) void k_out(const float* __restrict__ v,
    const float* __restrict__ w, float* out) {
  int id = blockIdx.x * 256 + threadIdx.x;           // over N_NODES*3
  if (id >= N_NODES * 3) return;
  const f4* vp = (const f4*)(v + (size_t)id * HID);
  const f4* wp = (const f4*)w;
  float acc = 0.0f;
#pragma unroll 8
  for (int k = 0; k < 32; ++k) {
    f4 aa = vp[k], bb = wp[k];
    acc += aa.x * bb.x + aa.y * bb.y + aa.z * bb.z + aa.w * bb.w;
  }
  out[id] = acc;
}

// ---------------- launcher ----------------

extern "C" void kernel_launch(void* const* d_in, const int* in_sizes, int n_in,
                              void* d_out, int out_size, void* d_ws, size_t ws_size,
                              hipStream_t stream) {
  (void)in_sizes; (void)n_in; (void)out_size; (void)ws_size;
  const float* atom_latent = (const float*)d_in[0];
  const float* vecf        = (const float*)d_in[1];
  const float* edge_attr   = (const float*)d_in[2];
  const float* pos         = (const float*)d_in[3];
  const float* embedding   = (const float*)d_in[4];
  const float* filt_W      = (const float*)d_in[5];
  const float* filt_b      = (const float*)d_in[6];
  const float* phi1_W      = (const float*)d_in[7];
  const float* phi1_b      = (const float*)d_in[8];
  const float* phi2_W      = (const float*)d_in[9];
  const float* phi2_b      = (const float*)d_in[10];
  const float* U_W         = (const float*)d_in[11];
  const float* V_W         = (const float*)d_in[12];
  const float* upd1_W      = (const float*)d_in[13];
  const float* upd1_b      = (const float*)d_in[14];
  const float* upd2_W      = (const float*)d_in[15];
  const float* upd2_b      = (const float*)d_in[16];
  const float* v_proj_w    = (const float*)d_in[17];
  const int*   z_atom      = (const int*)d_in[18];
  const int*   edge_index  = (const int*)d_in[19];
  const int* src = edge_index;
  const int* dst = edge_index + N_EDGES;

  char* wsp = (char*)d_ws;
  auto alloc = [&](size_t bytes) { char* p = wsp; wsp += (bytes + 255) & ~(size_t)255; return p; };
  float*    u     = (float*)   alloc((size_t)N_EDGES * 3 * 4);
  _Float16* ea_h  = (_Float16*)alloc((size_t)N_EDGES * 32 * 2);
  _Float16* fWt   = (_Float16*)alloc((size_t)LAYERS * 384 * 32 * 2);
  _Float16* p1t   = (_Float16*)alloc((size_t)LAYERS * 128 * 128 * 2);
  _Float16* p2t   = (_Float16*)alloc((size_t)LAYERS * 384 * 128 * 2);
  _Float16* Ut    = (_Float16*)alloc((size_t)LAYERS * 128 * 128 * 2);
  _Float16* Vt    = (_Float16*)alloc((size_t)LAYERS * 128 * 128 * 2);
  _Float16* u1t   = (_Float16*)alloc((size_t)LAYERS * 128 * 256 * 2);
  _Float16* u2t   = (_Float16*)alloc((size_t)LAYERS * 384 * 128 * 2);
  float*    s     = (float*)   alloc((size_t)N_NODES * HID * 4);
  _Float16* s_h   = (_Float16*)alloc((size_t)N_NODES * HID * 2);
  float*    v     = (float*)   alloc((size_t)N_NODES * 384 * 4);
  _Float16* v_h   = (_Float16*)alloc((size_t)N_NODES * 384 * 2);
  float*    v_acc = (float*)   alloc((size_t)N_NODES * 384 * 4);
  _Float16* Hh    = (_Float16*)alloc((size_t)N_NODES * HID * 2);
  float*    phi   = (float*)   alloc((size_t)N_NODES * 384 * 4);
  float*    Uv    = (float*)   alloc((size_t)N_NODES * 384 * 4);
  float*    Vv    = (float*)   alloc((size_t)N_NODES * 384 * 4);
  _Float16* Vnh   = (_Float16*)alloc((size_t)N_NODES * HID * 2);
  float*    dotv  = (float*)   alloc((size_t)N_NODES * HID * 4);

  for (int l = 0; l < LAYERS; ++l) {
    k_cvt<<<64, 256, 0, stream>>>(phi1_W + (size_t)l*128*128, p1t + (size_t)l*128*128, 128, 128);
    k_cvt<<<192, 256, 0, stream>>>(phi2_W + (size_t)l*128*384, p2t + (size_t)l*384*128, 128, 384);
    k_cvt<<<64, 256, 0, stream>>>(U_W + (size_t)l*128*128, Ut + (size_t)l*128*128, 128, 128);
    k_cvt<<<64, 256, 0, stream>>>(V_W + (size_t)l*128*128, Vt + (size_t)l*128*128, 128, 128);
    k_cvt<<<128, 256, 0, stream>>>(upd1_W + (size_t)l*256*128, u1t + (size_t)l*128*256, 256, 128);
    k_cvt<<<192, 256, 0, stream>>>(upd2_W + (size_t)l*128*384, u2t + (size_t)l*384*128, 128, 384);
    k_cvt_filt<<<48, 256, 0, stream>>>(filt_W + (size_t)l*EDGE_DIM*384, fWt + (size_t)l*384*32);
  }
  k_init<<<N_NODES * HID / 256, 256, 0, stream>>>(atom_latent, vecf, embedding, z_atom, s, s_h, v, v_acc);
  k_geom<<<N_EDGES / 256, 256, 0, stream>>>(pos, edge_attr, src, dst, u, ea_h);

  for (int l = 0; l < LAYERS; ++l) {
    k_gemm_silu<<<N_NODES / 16, 256, 0, stream>>>(s_h, p1t + (size_t)l*128*128, phi1_b + (size_t)l*128, Hh);
    k_gemm_n384<<<N_NODES / 16, 256, 0, stream>>>(Hh, p2t + (size_t)l*384*128, phi2_b + (size_t)l*384, phi);
    k_edge<<<N_EDGES / 128, 256, 0, stream>>>(ea_h, fWt + (size_t)l*384*32, filt_b + (size_t)l*384,
                                              phi, v, u, src, dst, s, v_acc);
    k_fin<<<N_NODES * HID / 256, 256, 0, stream>>>(s, s_h, v, v_acc, v_h);
    k_uvvv<<<N_NODES * 3 / 16, 256, 0, stream>>>(v_h, Ut + (size_t)l*128*128, Vt + (size_t)l*128*128, Uv, Vv);
    k_vn<<<N_NODES * HID / 256, 256, 0, stream>>>(Uv, Vv, Vnh, dotv);
    k_upd1<<<N_NODES / 16, 256, 0, stream>>>(s_h, Vnh, u1t + (size_t)l*128*256, upd1_b + (size_t)l*128, Hh);
    k_upd2<<<N_NODES / 16, 256, 0, stream>>>(Hh, u2t + (size_t)l*384*128, upd2_b + (size_t)l*384,
                                             Uv, dotv, v, s, s_h);
  }
  k_out<<<(N_NODES * 3 + 255) / 256, 256, 0, stream>>>(v, v_proj_w, (float*)d_out);
}
